// SyntaxSemanticsProbe_90993177133170
// MI455X (gfx1250) — matrix operand order repaired and checked by reading, not compiled, not run on hardware
//
#include <hip/hip_runtime.h>

// ---------------------------------------------------------------------------
// Problem constants (match reference setup_inputs)
// ---------------------------------------------------------------------------
constexpr int BB   = 8;
constexpr int NN   = 1024;
constexpr int DD   = 768;
constexpr int SRr  = 64;
constexpr int MRr  = 128;
constexpr int PRr  = SRr + MRr;   // 192
constexpr int ROWS = BB * NN;     // 8192
constexpr float EPSf = 1e-7f;
constexpr float MAXN = 1.0f - 1e-5f;

typedef __attribute__((ext_vector_type(16))) _Float16 v16h;
typedef __attribute__((ext_vector_type(8)))  float    v8f;

#define DEV __device__ __forceinline__

// ---------------------------------------------------------------------------
// WMMA f16 fragment loaders (row-major source, 16-byte vector loads)
// A: 16x32 (MxK).  lane<16: M=lane,    halves 0-7 -> K=k0+0..7,  8-15 -> K=k0+16..23
//                  lane>=16: M=lane-16, halves 0-7 -> K=k0+8..15, 8-15 -> K=k0+24..31
// B: 32x16 (KxN).  lane<16: N=lane,    halves 0-7 -> K=k0+0..7,  8-15 -> K=k0+16..23
//                  lane>=16: N=lane-16, halves 0-7 -> K=k0+8..15, 8-15 -> K=k0+24..31
// For C = X * Y^T both operands are contiguous row reads of X / Y.
// ---------------------------------------------------------------------------
union HFrag { v16h v; uint4 q[2]; };

DEV v16h load_fragA(const _Float16* X, int ld, int row, int k0, int lane) {
  const int kb = (lane >> 4) << 3;                 // 0 or 8
  const _Float16* p = X + (size_t)row * ld + k0 + kb;
  HFrag f;
  f.q[0] = *reinterpret_cast<const uint4*>(p);      // K = k0+kb .. +7
  f.q[1] = *reinterpret_cast<const uint4*>(p + 16); // K = k0+kb+16 .. +23
  return f.v;
}

DEV v16h load_fragB(const _Float16* X, int ld, int row, int k0, int lane) {
  const int kb = (lane >> 4) << 3;                 // 0 or 8
  const _Float16* p = X + (size_t)row * ld + k0 + kb;
  HFrag f;
  f.q[0] = *reinterpret_cast<const uint4*>(p);
  f.q[1] = *reinterpret_cast<const uint4*>(p + 16);
  return f.v;
}

DEV float waveRedAdd(float v) {
  #pragma unroll
  for (int m = 16; m >= 1; m >>= 1) v += __shfl_xor(v, m, 32);
  return v;
}

// ---------------------------------------------------------------------------
// 1) f32 -> f16 conversion of embeddings and concatenated weights [192x768]
// ---------------------------------------------------------------------------
__global__ void k_convert(const float* __restrict__ emb,
                          const float* __restrict__ wsyn,
                          const float* __restrict__ wsem,
                          _Float16* __restrict__ embH,
                          _Float16* __restrict__ wH) {
  const int tid    = blockIdx.x * blockDim.x + threadIdx.x;
  const int stride = gridDim.x * blockDim.x;
  const int nEmb = ROWS * DD;
  for (int i = tid; i < nEmb; i += stride) embH[i] = (_Float16)emb[i];
  const int nSyn = SRr * DD;
  for (int i = tid; i < nSyn; i += stride) wH[i] = (_Float16)wsyn[i];
  const int nSem = MRr * DD;
  for (int i = tid; i < nSem; i += stride) wH[nSyn + i] = (_Float16)wsem[i];
}

// ---------------------------------------------------------------------------
// 2) Projection GEMM: P[8192 x 192] = embH[8192 x 768] * wH[192 x 768]^T
//    One 16x16 tile per wave, 24 WMMA k-steps, f32 accumulate.
// ---------------------------------------------------------------------------
__global__ void __launch_bounds__(256) k_proj(const _Float16* __restrict__ embH,
                                              const _Float16* __restrict__ wH,
                                              float* __restrict__ P) {
  const int lane = threadIdx.x & 31;
  const int wave = threadIdx.x >> 5;
  const int tile = blockIdx.x * 8 + wave;          // 6144 tiles
  const int tm = tile / (PRr / 16);
  const int tn = tile % (PRr / 16);
  const int rowA = tm * 16 + (lane & 15);
  const int rowB = tn * 16 + (lane & 15);
  v8f acc = {};
  #pragma unroll 4
  for (int k0 = 0; k0 < DD; k0 += 32) {
    v16h a = load_fragA(embH, DD, rowA, k0, lane);
    v16h b = load_fragB(wH,   DD, rowB, k0, lane);
    acc = __builtin_amdgcn_wmma_f32_16x16x32_f16(false, a, false, b,
                                                 (short)0, acc, false, false);
  }
  const int hi  = lane >> 4;
  const int col = tn * 16 + (lane & 15);
  #pragma unroll
  for (int r = 0; r < 8; ++r) {
    const int row = tm * 16 + r + 8 * hi;          // C/D layout: VGPR r -> M=r / r+8
    P[(size_t)row * PRr + col] = acc[r];
  }
}

// ---------------------------------------------------------------------------
// 3) Per-row nonlinear stage (one wave per row; M staged in LDS per block):
//    syntax: expmap0 -> mobius_matvec -> quantize f16 + sq from quantized row
//    semantic: L2-normalize 128-dim row -> quantize f16
// ---------------------------------------------------------------------------
__global__ void __launch_bounds__(256) k_hyper(const float* __restrict__ P,
                                               const float* __restrict__ Mt,
                                               _Float16* __restrict__ synH,
                                               _Float16* __restrict__ semH,
                                               float* __restrict__ sqArr) {
  __shared__ float Msh[SRr * SRr];   // 16 KB: mobius_transform, row-major [s][r]
  __shared__ float pbuf[8][SRr];     // per-wave expmap0 result
  for (int i = threadIdx.x; i < SRr * SRr; i += blockDim.x) Msh[i] = Mt[i];
  __syncthreads();

  const int lane = threadIdx.x & 31;
  const int wave = threadIdx.x >> 5;
  const int row  = blockIdx.x * 8 + wave;          // 8192 rows
  const float* prow = P + (size_t)row * PRr;

  // ---- expmap0: p = tanh(||u||) * u / ||u|| (2 components per lane) ----
  const float u0 = prow[lane], u1 = prow[lane + 32];
  float n  = sqrtf(waveRedAdd(u0 * u0 + u1 * u1));
  n = fmaxf(n, EPSf);
  const float t = tanhf(n) / n;
  const float p0 = t * u0, p1 = t * u1;
  pbuf[wave][lane] = p0;
  pbuf[wave][lane + 32] = p1;
  const float xn = fminf(fmaxf(sqrtf(waveRedAdd(p0 * p0 + p1 * p1)), EPSf), MAXN);
  __syncthreads();

  // ---- mobius matvec: mx_s = sum_r p_r * M[s][r] ----
  const int s0 = lane, s1 = lane + 32;
  float mx0 = 0.f, mx1 = 0.f;
  #pragma unroll 8
  for (int r = 0; r < SRr; ++r) {
    const float pr = pbuf[wave][r];
    mx0 = fmaf(pr, Msh[s0 * SRr + r], mx0);
    mx1 = fmaf(pr, Msh[s1 * SRr + r], mx1);
  }
  const float mxn = fmaxf(sqrtf(waveRedAdd(mx0 * mx0 + mx1 * mx1)), EPSf);
  const float sc  = tanhf(mxn / xn * atanhf(xn)) / mxn;

  // quantize; compute sq from the quantized values so d^2 matches the WMMA gram
  const _Float16 h0 = (_Float16)(sc * mx0);
  const _Float16 h1 = (_Float16)(sc * mx1);
  synH[(size_t)row * SRr + lane]      = h0;
  synH[(size_t)row * SRr + lane + 32] = h1;
  const float f0 = (float)h0, f1 = (float)h1;
  const float sq = waveRedAdd(f0 * f0 + f1 * f1);
  if (lane == 0) sqArr[row] = fminf(sq, MAXN * MAXN);

  // ---- semantic branch: normalize 128-dim row (4 comps per lane) ----
  float v[4]; float sn2 = 0.f;
  #pragma unroll
  for (int c = 0; c < 4; ++c) {
    v[c] = prow[SRr + lane * 4 + c];
    sn2 += v[c] * v[c];
  }
  sn2 = waveRedAdd(sn2);
  const float inv = 1.0f / fmaxf(sqrtf(sn2), 1e-12f);
  #pragma unroll
  for (int c = 0; c < 4; ++c)
    semH[(size_t)row * MRr + lane * 4 + c] = (_Float16)(v[c] * inv);
}

// ---------------------------------------------------------------------------
// 4) Gram + epilogues. One 16x16 (i,j) tile per wave per batch:
//    syn gram (K=64, 2 WMMA) -> Poincare distance;  sem gram (K=128, 4 WMMA)
// ---------------------------------------------------------------------------
__global__ void __launch_bounds__(256) k_gram(const _Float16* __restrict__ synH,
                                              const _Float16* __restrict__ semH,
                                              const float* __restrict__ sqArr,
                                              float* __restrict__ outSyn,
                                              float* __restrict__ outSem) {
  const int lane = threadIdx.x & 31;
  const int wave = threadIdx.x >> 5;
  const int tile = blockIdx.x * 8 + wave;          // 8 * 64 * 64 = 32768 tiles
  const int b  = tile >> 12;
  const int rm = tile & 4095;
  const int ti = rm >> 6, tj = rm & 63;
  const int gi = b * NN + ti * 16 + (lane & 15);   // A-operand row
  const int gj = b * NN + tj * 16 + (lane & 15);   // B-operand row (X^T column)

  v8f accS = {};
  #pragma unroll
  for (int k0 = 0; k0 < SRr; k0 += 32) {
    v16h a  = load_fragA(synH, SRr, gi, k0, lane);
    v16h bb = load_fragB(synH, SRr, gj, k0, lane);
    accS = __builtin_amdgcn_wmma_f32_16x16x32_f16(false, a, false, bb,
                                                  (short)0, accS, false, false);
  }
  v8f accM = {};
  #pragma unroll
  for (int k0 = 0; k0 < MRr; k0 += 32) {
    v16h a  = load_fragA(semH, MRr, gi, k0, lane);
    v16h bb = load_fragB(semH, MRr, gj, k0, lane);
    accM = __builtin_amdgcn_wmma_f32_16x16x32_f16(false, a, false, bb,
                                                  (short)0, accM, false, false);
  }

  const int   hi   = lane >> 4;
  const float sqj  = sqArr[b * NN + tj * 16 + (lane & 15)];
  const float omj  = 1.0f - sqj;
  const size_t obase = (size_t)b * NN * NN;
  #pragma unroll
  for (int r = 0; r < 8; ++r) {
    const int   irow = ti * 16 + r + 8 * hi;
    const float sqi  = sqArr[b * NN + irow];
    const float g    = accS[r];
    const float d2   = fmaxf(sqi + sqj - 2.0f * g, 0.0f);
    const float den  = fmaxf((1.0f - sqi) * omj, EPSf);
    const float arg  = fmaxf(1.0f + 2.0f * d2 / den, 1.0f + EPSf);
    const size_t o = obase + (size_t)irow * NN + tj * 16 + (lane & 15);
    outSyn[o] = acoshf(arg);
    outSem[o] = accM[r];
  }
}

// ---------------------------------------------------------------------------
// Launch
// ---------------------------------------------------------------------------
extern "C" void kernel_launch(void* const* d_in, const int* in_sizes, int n_in,
                              void* d_out, int out_size, void* d_ws, size_t ws_size,
                              hipStream_t stream) {
  (void)in_sizes; (void)n_in; (void)out_size; (void)ws_size;
  const float* emb  = (const float*)d_in[0];   // [8,1024,768]
  const float* wsyn = (const float*)d_in[1];   // [64,768]
  const float* wsem = (const float*)d_in[2];   // [128,768]
  const float* Mt   = (const float*)d_in[3];   // [64,64]

  float* outSyn = (float*)d_out;                       // [8,1024,1024]
  float* outSem = outSyn + (size_t)BB * NN * NN;       // [8,1024,1024]

  char* ws = (char*)d_ws;
  size_t off = 0;
  auto take = [&](size_t bytes) -> void* {
    void* p = (void*)(ws + off);
    off += (bytes + 255) & ~(size_t)255;
    return p;
  };
  _Float16* embH = (_Float16*)take((size_t)ROWS * DD  * 2);   // 12.6 MB
  _Float16* wH   = (_Float16*)take((size_t)PRr  * DD  * 2);   // 0.3 MB
  float*    P    = (float*)   take((size_t)ROWS * PRr * 4);   // 6.3 MB
  _Float16* synH = (_Float16*)take((size_t)ROWS * SRr * 2);   // 1.0 MB
  _Float16* semH = (_Float16*)take((size_t)ROWS * MRr * 2);   // 2.1 MB
  float*    sqA  = (float*)   take((size_t)ROWS * 4);         // 32 KB

  k_convert<<<512, 256, 0, stream>>>(emb, wsyn, wsem, embH, wH);
  k_proj  <<<(ROWS / 16) * (PRr / 16) / 8, 256, 0, stream>>>(embH, wH, P);     // 768 blocks
  k_hyper <<<ROWS / 8, 256, 0, stream>>>(P, Mt, synH, semH, sqA);              // 1024 blocks
  k_gram  <<<BB * (NN / 16) * (NN / 16) / 8, 256, 0, stream>>>(synH, semH, sqA,
                                                               outSyn, outSem); // 4096 blocks
}
